// GroupLocalSL2_78924319031924
// MI455X (gfx1250) — matrix operand order repaired and checked by reading, not compiled, not run on hardware
//
#include <hip/hip_runtime.h>

// ---------------------------------------------------------------------------
// GroupLocalSL2 on MI455X (gfx1250, wave32, WMMA).
//
// out(b,o,g,h,w) = bias[o] +
//   sum_{c,gf,kh,kw} x[b,c,idx[g,gf],h+kh,w+kw] * weight[o,c,gf,kh,kw]
// B=8 C=32 O=64 G_IN=33 G_OUT=15 G_F=7 X=Y=64 K=5 -> out (8,64,15,60,60).
//
// Per (b,g): GEMM M=64 x N=3600 x K=5600.  fp32 via bf16 hi/lo split
// (3x V_WMMA_F32_16X16X32_BF16 per tile: ~16 mantissa bits).
// Block = 2 output rows (N=128, 120 valid) x all 64 out channels.
// K->x-offset decomposition precomputed per (g,k) (L2-resident table).
// Fast path: x pre-split once into packed u32 {lo16,hi16}; hot-loop fill is
// pure v_perm_b32 merges.  LDS B tiles are ping-pong double-buffered ->
// one barrier per K-chunk.
// ---------------------------------------------------------------------------

typedef __attribute__((ext_vector_type(16))) __bf16 v16bf;
typedef __attribute__((ext_vector_type(8)))  __bf16 v8bf;
typedef __attribute__((ext_vector_type(8)))  float  v8f;
typedef unsigned int u32;

#define C_CH   32
#define O_CH   64
#define G_IN   33
#define G_OUT  15
#define G_F    7
#define KSZ    5
#define IMG    64
#define OUTW   60
#define KDIM   5600              // C_CH * G_F * KSZ * KSZ
#define KCHUNK 32
#define NCHUNK (KDIM / KCHUNK)   // 175
#define NTILE  128               // 2 output rows x 64 (60 valid) cols
#define XELEMS ((size_t)8 * C_CH * G_IN * IMG * IMG)   // 34,603,008

union FragBF { v16bf v; v8bf h[2]; };
union BF16U  { __bf16 b; unsigned short u; };

static __device__ __forceinline__ v8f wmma_bf16(const FragBF& a, const FragBF& b, v8f c) {
    return __builtin_amdgcn_wmma_f32_16x16x32_bf16(
        false, a.v, false, b.v, (short)0, c, false, false);
}

// ---------------- prep 1: split weight into bf16 hi/lo ---------------------
__global__ __launch_bounds__(256) void GroupLocalSL2_wsplit(
    const float* __restrict__ w, __bf16* __restrict__ w_hi, __bf16* __restrict__ w_lo) {
    int i = blockIdx.x * 256 + threadIdx.x;
    if (i < O_CH * KDIM) {
        float v  = w[i];
        __bf16 h = (__bf16)v;
        w_hi[i]  = h;
        w_lo[i]  = (__bf16)(v - (float)h);
    }
}

// ---------------- prep 2: per-(g,k) x-offset table -------------------------
__global__ __launch_bounds__(256) void GroupLocalSL2_offsets(
    const int* __restrict__ idx, int* __restrict__ offT) {
    int i = blockIdx.x * 256 + threadIdx.x;
    if (i < G_OUT * KDIM) {
        int g   = i / KDIM;
        int k   = i - g * KDIM;
        int kw  = k % KSZ;
        int t5  = k / KSZ;
        int kh  = t5 % KSZ;
        int t25 = t5 / KSZ;
        int gf  = t25 % G_F;
        int c   = t25 / G_F;
        int gin = idx[g * G_F + gf];
        offT[i] = ((c * G_IN + gin) << 12) + (kh << 6) + kw;
    }
}

// ---------------- prep 3: split x into packed {lo16,hi16} u32 --------------
__global__ __launch_bounds__(256) void GroupLocalSL2_xpack(
    const float* __restrict__ x, u32* __restrict__ xp) {
    size_t i = (size_t)blockIdx.x * 256 + threadIdx.x;
    if (i < XELEMS) {
        float v = x[i];
        BF16U hi, lo;
        hi.b = (__bf16)v;
        lo.b = (__bf16)(v - (float)hi.b);
        xp[i] = ((u32)lo.u << 16) | (u32)hi.u;
    }
}

// ===========================================================================
// Fast-path main kernel: x pre-split (packed).  grid = (30 hb, 15 g, 8 b).
// ===========================================================================
__global__ __launch_bounds__(256) void GroupLocalSL2_conv_pk(
    const u32*    __restrict__ xp,
    const __bf16* __restrict__ w_hi,
    const __bf16* __restrict__ w_lo,
    const int*    __restrict__ offT,
    const float*  __restrict__ bias,
    float*        __restrict__ out) {

    __shared__ __align__(16) __bf16 sBh[2][NTILE][KCHUNK];   // ping-pong
    __shared__ __align__(16) __bf16 sBl[2][NTILE][KCHUNK];

    const int h0  = blockIdx.x * 2;
    const int g   = blockIdx.y;
    const int b   = blockIdx.z;
    const int tid  = threadIdx.x;
    const int lane = tid & 31;
    const int wave = tid >> 5;

    const int  wcol  = tid >> 2;
    const int  kl0   = (tid & 3) * 8;
    const bool valid = (wcol < OUTW);
    const int* offG  = offT + g * KDIM + kl0;
    const u32* xp0 = xp + (((size_t)b * C_CH * G_IN) << 12) + (h0 << 6) + wcol;
    const u32* xp1 = xp0 + IMG;

    const int mtile = wave & 3;
    const int nbase = (wave >> 2) * 4;

    const int arow  = 16 * mtile + (lane & 15);
    const int akoff = (lane >= 16) ? 16 : 0;
    const __bf16* whRow = w_hi + arow * KDIM + akoff;
    const __bf16* wlRow = w_lo + arow * KDIM + akoff;

    const int bcol = (lane & 15);
    const int bk   = (lane >= 16) ? 16 : 0;

    auto loadX = [&](int kc, u32* a0, u32* a1) {
        const int4* p = (const int4*)(offG + kc * KCHUNK);
        int4 oa = p[0], ob = p[1];
        int offs[8] = {oa.x, oa.y, oa.z, oa.w, ob.x, ob.y, ob.z, ob.w};
        #pragma unroll
        for (int j = 0; j < 8; ++j) {
            a0[j] = valid ? xp0[offs[j]] : 0u;
            a1[j] = valid ? xp1[offs[j]] : 0u;
        }
    };
    // hi-dword = {e.b0,e.b1,o.b0,o.b1} ; lo-dword = {e.b2,e.b3,o.b2,o.b3}
    // v_perm_b32: sel bytes 0-3 pick S1 (=e), 4-7 pick S0 (=o)
    auto storeLDS = [&](int buf, const u32* a0, const u32* a1) {
        u32 h0v[4], l0v[4], h1v[4], l1v[4];
        #pragma unroll
        for (int d = 0; d < 4; ++d) {
            u32 e0 = a0[2 * d], o0 = a0[2 * d + 1];
            h0v[d] = __builtin_amdgcn_perm(o0, e0, 0x05040100u);
            l0v[d] = __builtin_amdgcn_perm(o0, e0, 0x07060302u);
            u32 e1 = a1[2 * d], o1 = a1[2 * d + 1];
            h1v[d] = __builtin_amdgcn_perm(o1, e1, 0x05040100u);
            l1v[d] = __builtin_amdgcn_perm(o1, e1, 0x07060302u);
        }
        *(uint4*)&sBh[buf][wcol     ][kl0] = *(const uint4*)h0v;
        *(uint4*)&sBl[buf][wcol     ][kl0] = *(const uint4*)l0v;
        *(uint4*)&sBh[buf][wcol + 64][kl0] = *(const uint4*)h1v;
        *(uint4*)&sBl[buf][wcol + 64][kl0] = *(const uint4*)l1v;
    };

    v8f acc[4] = {};
    {
        u32 xv0[8], xv1[8];
        loadX(0, xv0, xv1);
        storeLDS(0, xv0, xv1);
    }
    __syncthreads();

    #pragma unroll 2
    for (int kc = 0; kc < NCHUNK; ++kc) {
        const int cur = kc & 1;
        u32 nx0[8], nx1[8];
        if (kc + 1 < NCHUNK) loadX(kc + 1, nx0, nx1);

        FragBF ah, al;
        {
            const __bf16* p = whRow + kc * KCHUNK;
            ah.h[0] = *(const v8bf*)(p);
            ah.h[1] = *(const v8bf*)(p + 8);
            const __bf16* q = wlRow + kc * KCHUNK;
            al.h[0] = *(const v8bf*)(q);
            al.h[1] = *(const v8bf*)(q + 8);
        }

        #pragma unroll
        for (int t = 0; t < 4; ++t) {
            const int cc = (nbase + t) * 16 + bcol;
            FragBF bh, bl;
            bh.h[0] = *(const v8bf*)&sBh[cur][cc][bk];
            bh.h[1] = *(const v8bf*)&sBh[cur][cc][bk + 8];
            bl.h[0] = *(const v8bf*)&sBl[cur][cc][bk];
            bl.h[1] = *(const v8bf*)&sBl[cur][cc][bk + 8];
            acc[t] = wmma_bf16(ah, bh, acc[t]);
            acc[t] = wmma_bf16(ah, bl, acc[t]);
            acc[t] = wmma_bf16(al, bh, acc[t]);
        }

        // store next chunk into the other buffer: no reader conflict,
        // single barrier publishes it for the next iteration
        if (kc + 1 < NCHUNK) storeLDS(cur ^ 1, nx0, nx1);
        __syncthreads();
    }

    const int obase = 16 * mtile + ((lane >= 16) ? 8 : 0);
    #pragma unroll
    for (int r = 0; r < 8; ++r) {
        const int o  = obase + r;
        const float bv = bias[o];
        #pragma unroll
        for (int t = 0; t < 4; ++t) {
            const int n   = (nbase + t) * 16 + (lane & 15);
            const int row = n >> 6;
            const int w   = n & 63;
            if (w < OUTW) {
                const size_t off = ((((size_t)b * O_CH + o) * G_OUT + g) * OUTW
                                    + (h0 + row)) * OUTW + w;
                out[off] = acc[t][r] + bv;
            }
        }
    }
}

// ===========================================================================
// Fallback main kernel (small workspace): convert fp32->bf16 in the fill.
// ===========================================================================
__global__ __launch_bounds__(256) void GroupLocalSL2_conv_cv(
    const float*  __restrict__ x,
    const __bf16* __restrict__ w_hi,
    const __bf16* __restrict__ w_lo,
    const int*    __restrict__ offT,
    const float*  __restrict__ bias,
    float*        __restrict__ out) {

    __shared__ __align__(16) __bf16 sBh[2][NTILE][KCHUNK];
    __shared__ __align__(16) __bf16 sBl[2][NTILE][KCHUNK];

    const int h0  = blockIdx.x * 2;
    const int g   = blockIdx.y;
    const int b   = blockIdx.z;
    const int tid  = threadIdx.x;
    const int lane = tid & 31;
    const int wave = tid >> 5;

    const int  wcol  = tid >> 2;
    const int  kl0   = (tid & 3) * 8;
    const bool valid = (wcol < OUTW);
    const int* offG  = offT + g * KDIM + kl0;
    const float* xb0 = x + (((size_t)b * C_CH * G_IN) << 12) + (h0 << 6) + wcol;
    const float* xb1 = xb0 + IMG;

    const int mtile = wave & 3;
    const int nbase = (wave >> 2) * 4;

    const int arow  = 16 * mtile + (lane & 15);
    const int akoff = (lane >= 16) ? 16 : 0;
    const __bf16* whRow = w_hi + arow * KDIM + akoff;
    const __bf16* wlRow = w_lo + arow * KDIM + akoff;

    const int bcol = (lane & 15);
    const int bk   = (lane >= 16) ? 16 : 0;

    auto loadX = [&](int kc, float* v0, float* v1) {
        const int4* p = (const int4*)(offG + kc * KCHUNK);
        int4 oa = p[0], ob = p[1];
        int offs[8] = {oa.x, oa.y, oa.z, oa.w, ob.x, ob.y, ob.z, ob.w};
        #pragma unroll
        for (int j = 0; j < 8; ++j) {
            v0[j] = valid ? xb0[offs[j]] : 0.0f;
            v1[j] = valid ? xb1[offs[j]] : 0.0f;
        }
    };
    auto storeLDS = [&](int buf, const float* v0, const float* v1) {
        v8bf h0v, l0v, h1v, l1v;
        #pragma unroll
        for (int j = 0; j < 8; ++j) {
            __bf16 a = (__bf16)v0[j];  h0v[j] = a;  l0v[j] = (__bf16)(v0[j] - (float)a);
            __bf16 c = (__bf16)v1[j];  h1v[j] = c;  l1v[j] = (__bf16)(v1[j] - (float)c);
        }
        *(v8bf*)&sBh[buf][wcol     ][kl0] = h0v;
        *(v8bf*)&sBl[buf][wcol     ][kl0] = l0v;
        *(v8bf*)&sBh[buf][wcol + 64][kl0] = h1v;
        *(v8bf*)&sBl[buf][wcol + 64][kl0] = l1v;
    };

    v8f acc[4] = {};
    {
        float xv0[8], xv1[8];
        loadX(0, xv0, xv1);
        storeLDS(0, xv0, xv1);
    }
    __syncthreads();

    #pragma unroll 2
    for (int kc = 0; kc < NCHUNK; ++kc) {
        const int cur = kc & 1;
        float nx0[8], nx1[8];
        if (kc + 1 < NCHUNK) loadX(kc + 1, nx0, nx1);

        FragBF ah, al;
        {
            const __bf16* p = whRow + kc * KCHUNK;
            ah.h[0] = *(const v8bf*)(p);
            ah.h[1] = *(const v8bf*)(p + 8);
            const __bf16* q = wlRow + kc * KCHUNK;
            al.h[0] = *(const v8bf*)(q);
            al.h[1] = *(const v8bf*)(q + 8);
        }

        #pragma unroll
        for (int t = 0; t < 4; ++t) {
            const int cc = (nbase + t) * 16 + bcol;
            FragBF bh, bl;
            bh.h[0] = *(const v8bf*)&sBh[cur][cc][bk];
            bh.h[1] = *(const v8bf*)&sBh[cur][cc][bk + 8];
            bl.h[0] = *(const v8bf*)&sBl[cur][cc][bk];
            bl.h[1] = *(const v8bf*)&sBl[cur][cc][bk + 8];
            acc[t] = wmma_bf16(ah, bh, acc[t]);
            acc[t] = wmma_bf16(ah, bl, acc[t]);
            acc[t] = wmma_bf16(al, bh, acc[t]);
        }

        if (kc + 1 < NCHUNK) storeLDS(cur ^ 1, nx0, nx1);
        __syncthreads();
    }

    const int obase = 16 * mtile + ((lane >= 16) ? 8 : 0);
    #pragma unroll
    for (int r = 0; r < 8; ++r) {
        const int o  = obase + r;
        const float bv = bias[o];
        #pragma unroll
        for (int t = 0; t < 4; ++t) {
            const int n   = (nbase + t) * 16 + (lane & 15);
            const int row = n >> 6;
            const int w   = n & 63;
            if (w < OUTW) {
                const size_t off = ((((size_t)b * O_CH + o) * G_OUT + g) * OUTW
                                    + (h0 + row)) * OUTW + w;
                out[off] = acc[t][r] + bv;
            }
        }
    }
}

// ------------------------------- launcher ----------------------------------
extern "C" void kernel_launch(void* const* d_in, const int* in_sizes, int n_in,
                              void* d_out, int out_size, void* d_ws, size_t ws_size,
                              hipStream_t stream) {
    const float* x      = (const float*)d_in[0];   // (8,32,33,64,64) f32
    const float* weight = (const float*)d_in[1];   // (64,32,7,5,5)   f32
    const float* bias   = (const float*)d_in[2];   // (64,)           f32
    const int*   idx    = (const int*)  d_in[3];   // (15,7)          i32
    float*       out    = (float*)d_out;           // (8,64,15,60,60) f32

    // ws layout: w_hi | w_lo (bf16 [64][5600]) | offT (int [15][5600]) | x_pk
    const size_t wElems   = (size_t)O_CH * KDIM;
    __bf16* w_hi = (__bf16*)d_ws;
    __bf16* w_lo = w_hi + wElems;
    int*    offT = (int*)(w_lo + wElems);
    u32*    x_pk = (u32*)(offT + (size_t)G_OUT * KDIM);
    const size_t needFast = 2 * wElems * 2 + (size_t)G_OUT * KDIM * 4
                          + XELEMS * 4 + 64;

    GroupLocalSL2_wsplit <<<(O_CH * KDIM + 255) / 256, 256, 0, stream>>>(weight, w_hi, w_lo);
    GroupLocalSL2_offsets<<<(G_OUT * KDIM + 255) / 256, 256, 0, stream>>>(idx, offT);

    dim3 grid(OUTW / 2, G_OUT, 8);   // 3600 workgroups
    if (ws_size >= needFast) {
        GroupLocalSL2_xpack<<<(unsigned)((XELEMS + 255) / 256), 256, 0, stream>>>(x, x_pk);
        GroupLocalSL2_conv_pk<<<grid, 256, 0, stream>>>(x_pk, w_hi, w_lo, offT, bias, out);
    } else {
        GroupLocalSL2_conv_cv<<<grid, 256, 0, stream>>>(x, w_hi, w_lo, offT, bias, out);
    }
}
